// OCAB_86655260164343
// MI455X (gfx1250) — matrix operand order repaired and checked by reading, not compiled, not run on hardware
//
#include <hip/hip_runtime.h>
#include <hip/hip_bf16.h>
#include <stdint.h>

// ---------------- types / helpers ----------------
typedef __attribute__((ext_vector_type(16))) __bf16 v16bf;
typedef __attribute__((ext_vector_type(8)))  float  v8f;

union Frag { v16bf v; __bf16 e[16]; uint4 q[2]; };

__device__ __forceinline__ __bf16 f2bf(float f){
  unsigned u = __builtin_bit_cast(unsigned, f);
  u += 0x7FFFu + ((u >> 16) & 1u);            // round to nearest even
  unsigned short h = (unsigned short)(u >> 16);
  return __builtin_bit_cast(__bf16, h);
}
__device__ __forceinline__ float bf2f(__bf16 b){
  unsigned short s = __builtin_bit_cast(unsigned short, b);
  unsigned u = ((unsigned)s) << 16;
  return __builtin_bit_cast(float, u);
}
__device__ __forceinline__ __bf16 us2bf(unsigned short s){
  return __builtin_bit_cast(__bf16, s);
}

#define C_    180
#define CP    192          // C padded to K-multiple of 32
#define NQKV  576          // 540 padded to 64-multiple
#define NPROJ 192          // 180 padded
#define NM1   384          // 360 padded
#define KM2   384
#define ROWS  131072       // B*H*W = 2*256*256
#define HDIM  30
#define SLD   584          // padded LDS row stride for S/P (f32 / bf16)
#define KLD   592          // padded LDS row stride for transposed K (bf16)

// ---------------- weight / bias pad-convert ----------------
__global__ void pad_w_kernel(const float* __restrict__ src, __bf16* __restrict__ dst,
                             int sR, int sC, int dR, int dC){
  int i = blockIdx.x*blockDim.x + threadIdx.x;
  if(i >= dR*dC) return;
  int r = i/dC, c = i%dC;
  dst[i] = f2bf((r < sR && c < sC) ? src[r*sC + c] : 0.f);
}
__global__ void pad_b_kernel(const float* __restrict__ src, float* __restrict__ dst,
                             int sC, int dC){
  int i = blockIdx.x*blockDim.x + threadIdx.x;
  if(i >= dC) return;
  dst[i] = (i < sC) ? src[i] : 0.f;
}
// bias[h][qpos][key] = rpb_table[rpi[qpos][key]][h]
__global__ void bias_kernel(const int* __restrict__ rpi, const float* __restrict__ rpb,
                            float* __restrict__ biasT){
  int i = blockIdx.x*blockDim.x + threadIdx.x;
  if(i >= 6*256*576) return;
  int h = i / (256*576);
  int rem = i % (256*576);
  biasT[i] = rpb[rpi[rem]*6 + h];
}

// ---------------- LayerNorm -> padded bf16 ----------------
__global__ void ln_kernel(const float* __restrict__ x, const float* __restrict__ g,
                          const float* __restrict__ bta, __bf16* __restrict__ out){
  const long row = blockIdx.x;
  const float* xr = x + row*(long)C_;
  __shared__ float red[64];
  int t = threadIdx.x;
  float s = 0.f;
  for(int c = t; c < C_; c += 64) s += xr[c];
  red[t] = s; __syncthreads();
  for(int o = 32; o > 0; o >>= 1){ if(t < o) red[t] += red[t+o]; __syncthreads(); }
  float mu = red[0]*(1.0f/C_);
  __syncthreads();
  float s2 = 0.f;
  for(int c = t; c < C_; c += 64){ float d = xr[c]-mu; s2 += d*d; }
  red[t] = s2; __syncthreads();
  for(int o = 32; o > 0; o >>= 1){ if(t < o) red[t] += red[t+o]; __syncthreads(); }
  float rinv = rsqrtf(red[0]*(1.0f/C_) + 1e-5f);
  for(int c = t; c < CP; c += 64){
    float v = (c < C_) ? (xr[c]-mu)*rinv*g[c] + bta[c] : 0.f;
    out[row*(long)CP + c] = f2bf(v);
  }
}

// ---------------- bf16 WMMA GEMM, 128x64 block tile, 8 waves ----------------
// EPI 0: bf16 out = acc+bias       (qkv)
// EPI 1: f32 out[realN] = acc+bias+extra   (proj + residual)
// EPI 2: bf16 out = gelu(acc+bias) (mlp1)
// EPI 3: f32 out[realN] = acc+bias+extra   (mlp2 + residual)
template<int EPI>
__global__ void gemm_wmma(const __bf16* __restrict__ A, const __bf16* __restrict__ B,
                          const float* __restrict__ bias, const float* __restrict__ extra,
                          float* __restrict__ outF, __bf16* __restrict__ outB,
                          int K, int N, int realN){
  const int lane = threadIdx.x & 31;
  const int wv   = threadIdx.x >> 5;
  const int half = lane >> 4;
  const int l16  = lane & 15;
  const long m0  = (long)blockIdx.x*128 + (wv>>1)*32;
  const int  n0  = blockIdx.y*64 + (wv&1)*32;

  v8f acc[2][2];
  #pragma unroll
  for(int i=0;i<2;i++)
    #pragma unroll
    for(int j=0;j<2;j++)
      #pragma unroll
      for(int v=0;v<8;v++) acc[i][j][v] = 0.f;

  for(int k0 = 0; k0 < K; k0 += 32){
    Frag a[2], bb[2];
    #pragma unroll
    for(int i=0;i<2;i++){
      const __bf16* ap = A + (m0 + i*16 + l16)*(long)K + k0 + half*8;
      a[i].q[0] = *(const uint4*)ap;
      a[i].q[1] = *(const uint4*)(ap + 16);
      if(k0 + 32 < K) __builtin_prefetch(ap + 32, 0, 1);   // global_prefetch_b8
    }
    #pragma unroll
    for(int j=0;j<2;j++){
      const __bf16* bp = B + (long)k0*N + n0 + j*16 + l16;
      #pragma unroll
      for(int s=0;s<16;s++) bb[j].e[s] = bp[(long)((s&7) + half*8 + (s>>3)*16)*N];
    }
    #pragma unroll
    for(int i=0;i<2;i++)
      #pragma unroll
      for(int j=0;j<2;j++)
        acc[i][j] = __builtin_amdgcn_wmma_f32_16x16x32_bf16(
            false, a[i].v, false, bb[j].v, (short)0, acc[i][j], false, false);
  }

  #pragma unroll
  for(int i=0;i<2;i++)
  #pragma unroll
  for(int j=0;j<2;j++)
  #pragma unroll
  for(int v=0;v<8;v++){
    long r = m0 + i*16 + v + 8*half;
    int  c = n0 + j*16 + l16;
    float val = acc[i][j][v] + bias[c];
    if(EPI == 0){
      outB[r*(long)N + c] = f2bf(val);
    } else if(EPI == 2){
      float ge = 0.5f*val*(1.f + erff(val*0.70710678118f));
      outB[r*(long)N + c] = f2bf(ge);
    } else {
      if(c < realN) outF[r*(long)realN + c] = val + extra[r*(long)realN + c];
    }
  }
}

// ---------------- overlapping-window attention ----------------
// one block (256 thr, 8 waves) per (window, head); loops over 16 q-tiles.
// K/V staged to LDS exactly once per (window, head) -> 16x less staging traffic.
// LDS: Sl[16][SLD] f32 | Ol[16][32] f32 | Klt[32][KLD] bf16 (ch-major) |
//      Vl[576][32] bf16 | Pl[16][SLD] bf16
__global__ void attn_kernel(const __bf16* __restrict__ QKV,
                            const float* __restrict__ biasT,
                            __bf16* __restrict__ O){
  extern __shared__ char smem[];
  float*  Sl  = (float*)smem;                      // 16*SLD f32
  float*  Ol  = Sl + 16*SLD;                       // 16*32  f32
  __bf16* Klt = (__bf16*)(Ol + 16*32);             // 32*KLD bf16 (transposed)
  __bf16* Vl  = Klt + 32*KLD;                      // 576*32 bf16
  __bf16* Pl  = Vl + 576*32;                       // 16*SLD bf16

  const int tid = threadIdx.x, lane = tid & 31, wv = tid >> 5;
  const int half = lane >> 4, l16 = lane & 15;
  int bid  = blockIdx.x;                           // 512*6 blocks
  int head = bid % 6;
  int win  = bid / 6;
  int b  = win >> 8;
  int wh = (win >> 4) & 15;
  int ww = win & 15;

  // stage K (transposed) and V once: (key, channel-pair) dword global loads
  for(int idx = tid; idx < 576*16; idx += 256){
    int key = idx >> 4, cp = idx & 15;             // channels 2cp, 2cp+1
    int gy = wh*16 - 4 + key/24, gx = ww*16 - 4 + key%24;
    unsigned kk = 0u, vv = 0u;
    if(gy >= 0 && gy < 256 && gx >= 0 && gx < 256 && cp < 15){
      long rowq = (((long)b*256 + gy)*256 + gx)*(long)NQKV;
      kk = *(const unsigned*)(QKV + rowq + C_   + head*HDIM + 2*cp);
      vv = *(const unsigned*)(QKV + rowq + 2*C_ + head*HDIM + 2*cp);
    }
    *(unsigned*)&Vl[key*32 + 2*cp] = vv;
    Klt[(2*cp  )*KLD + key] = us2bf((unsigned short)(kk & 0xFFFFu));
    Klt[(2*cp+1)*KLD + key] = us2bf((unsigned short)(kk >> 16));
  }
  const float* bT = biasT + (long)head*(256*576);
  __syncthreads();

  for(int qt = 0; qt < 16; ++qt){
    // Q fragment (A layout: lane=row, K striped), pre-scaled by d^-1/2
    Frag qf;
    {
      const float scale = 0.18257418583505536f;    // 30^-0.5
      int p = qt*16 + l16;
      int y = wh*16 + (p >> 4), x = ww*16 + (p & 15);
      long rowq = (((long)b*256 + y)*256 + x)*(long)NQKV + head*HDIM;
      #pragma unroll
      for(int e2 = 0; e2 < 8; e2++){
        int c0 = half*8 + e2;
        int c1 = 16 + half*8 + e2;
        qf.e[e2]     = (c0 < HDIM) ? f2bf(scale*bf2f(QKV[rowq + c0])) : us2bf(0);
        qf.e[8 + e2] = (c1 < HDIM) ? f2bf(scale*bf2f(QKV[rowq + c1])) : us2bf(0);
      }
    }

    // S = q kT + bias ; 36 col-tiles across 8 waves
    for(int t = wv; t < 36; t += 8){
      // prefetch the 8 bias values first so they clause and overlap the WMMA
      float bvals[8];
      {
        const float* bp = bT + (qt*16 + 8*half)*576 + t*16 + l16;
        #pragma unroll
        for(int v = 0; v < 8; v++) bvals[v] = bp[v*576];
      }
      Frag kb;                                     // lane = key column; channels striped (table order)
      const __bf16* kp = &Klt[t*16 + l16];
      #pragma unroll
      for(int s = 0; s < 16; s++) kb.e[s] = kp[((s&7) + half*8 + (s>>3)*16)*KLD];
      v8f s;
      #pragma unroll
      for(int v = 0; v < 8; v++) s[v] = 0.f;
      s = __builtin_amdgcn_wmma_f32_16x16x32_bf16(false, qf.v, false, kb.v, (short)0, s, false, false);
      #pragma unroll
      for(int v = 0; v < 8; v++)
        Sl[(v + 8*half)*SLD + t*16 + l16] = s[v] + bvals[v];
    }
    __syncthreads();

    // softmax: 16 lanes per row (half-wave shuffles); also zero O accumulators
    {
      int r = tid >> 4, g = tid & 15;
      float mx = -3.0e38f;
      for(int c = g; c < 576; c += 16) mx = fmaxf(mx, Sl[r*SLD + c]);
      for(int o = 8; o > 0; o >>= 1) mx = fmaxf(mx, __shfl_xor(mx, o, 16));
      float sum = 0.f;
      for(int c = g; c < 576; c += 16){
        float e = __expf(Sl[r*SLD + c] - mx);
        Sl[r*SLD + c] = e; sum += e;
      }
      for(int o = 8; o > 0; o >>= 1) sum += __shfl_xor(sum, o, 16);
      float inv = 1.f/sum;
      for(int c = g; c < 576; c += 16) Pl[r*SLD + c] = f2bf(Sl[r*SLD + c]*inv);
    }
    for(int idx = tid; idx < 512; idx += 256) Ol[idx] = 0.f;
    __syncthreads();

    // O = P @ V ; 18 K-chunks across 8 waves, accumulate via LDS float atomics
    for(int kc = wv; kc < 18; kc += 8){
      Frag pf;                                     // lane = q row; key slices in table order
      const __bf16* pp = &Pl[l16*SLD + kc*32 + half*8];
      pf.q[0] = *(const uint4*)pp;
      pf.q[1] = *(const uint4*)(pp + 16);
      #pragma unroll
      for(int j = 0; j < 2; j++){
        Frag vb;                                   // lane = channel column; keys striped (table order)
        const __bf16* vp = &Vl[(kc*32)*32 + j*16 + l16];
        #pragma unroll
        for(int s = 0; s < 16; s++) vb.e[s] = vp[((s&7) + half*8 + (s>>3)*16)*32];
        v8f o;
        #pragma unroll
        for(int v = 0; v < 8; v++) o[v] = 0.f;
        o = __builtin_amdgcn_wmma_f32_16x16x32_bf16(false, pf.v, false, vb.v, (short)0, o, false, false);
        #pragma unroll
        for(int v = 0; v < 8; v++)
          atomicAdd(&Ol[(v + 8*half)*32 + j*16 + l16], o[v]);
      }
    }
    __syncthreads();

    // write 16x30 output tile (padded bf16 layout [ROWS, CP])
    for(int idx = tid; idx < 512; idx += 256){
      int r = idx >> 5, c = idx & 31;
      if(c < HDIM){
        int p = qt*16 + r;
        int y = wh*16 + (p >> 4), x = ww*16 + (p & 15);
        long row = ((long)b*256 + y)*256 + x;
        O[row*(long)CP + head*HDIM + c] = f2bf(Ol[idx]);
      }
    }
    __syncthreads();
  }
}

// ---------------- host driver ----------------
extern "C" void kernel_launch(void* const* d_in, const int* in_sizes, int n_in,
                              void* d_out, int out_size, void* d_ws, size_t ws_size,
                              hipStream_t stream){
  const float* x      = (const float*)d_in[0];
  const int*   rpi    = (const int*)  d_in[1];
  const float* qkv_w  = (const float*)d_in[2];
  const float* qkv_b  = (const float*)d_in[3];
  const float* proj_w = (const float*)d_in[4];
  const float* proj_b = (const float*)d_in[5];
  const float* rpb    = (const float*)d_in[6];
  const float* ln1_g  = (const float*)d_in[7];
  const float* ln1_b  = (const float*)d_in[8];
  const float* ln2_g  = (const float*)d_in[9];
  const float* ln2_b  = (const float*)d_in[10];
  const float* w1     = (const float*)d_in[11];
  const float* b1     = (const float*)d_in[12];
  const float* w2     = (const float*)d_in[13];
  const float* b2     = (const float*)d_in[14];
  float* out = (float*)d_out;

  char* ws = (char*)d_ws;
  size_t off = 0;
  auto carve = [&](size_t bytes)->char*{
    char* p = ws + off; off = (off + bytes + 255) & ~(size_t)255; return p;
  };
  __bf16* hA    = (__bf16*)carve((size_t)ROWS*CP*2);
  __bf16* Wqkv  = (__bf16*)carve((size_t)CP*NQKV*2);
  float*  Bqkv  = (float*) carve((size_t)NQKV*4);
  __bf16* Wproj = (__bf16*)carve((size_t)CP*NPROJ*2);
  float*  Bproj = (float*) carve((size_t)NPROJ*4);
  __bf16* W1    = (__bf16*)carve((size_t)CP*NM1*2);
  float*  B1    = (float*) carve((size_t)NM1*4);
  __bf16* W2    = (__bf16*)carve((size_t)KM2*NPROJ*2);
  float*  B2    = (float*) carve((size_t)NPROJ*4);
  float*  BiasT = (float*) carve((size_t)6*256*576*4);
  __bf16* QKV   = (__bf16*)carve((size_t)ROWS*NQKV*2);
  __bf16* Obuf  = (__bf16*)carve((size_t)ROWS*CP*2);
  float*  X2    = (float*) carve((size_t)ROWS*C_*4);
  __bf16* H2    = (__bf16*)carve((size_t)ROWS*CP*2);
  __bf16* M1    = (__bf16*)carve((size_t)ROWS*NM1*2);

  // weights / biases -> padded bf16 / f32 ; bias table gather
  pad_w_kernel<<<(CP*NQKV + 255)/256, 256, 0, stream>>>(qkv_w,  Wqkv,  C_, 540, CP, NQKV);
  pad_w_kernel<<<(CP*NPROJ + 255)/256, 256, 0, stream>>>(proj_w, Wproj, C_, 180, CP, NPROJ);
  pad_w_kernel<<<(CP*NM1 + 255)/256, 256, 0, stream>>>(w1,      W1,    C_, 360, CP, NM1);
  pad_w_kernel<<<(KM2*NPROJ + 255)/256, 256, 0, stream>>>(w2,    W2,   360, 180, KM2, NPROJ);
  pad_b_kernel<<<(NQKV + 255)/256, 256, 0, stream>>>(qkv_b,  Bqkv,  540, NQKV);
  pad_b_kernel<<<1, 256, 0, stream>>>(proj_b, Bproj, 180, NPROJ);
  pad_b_kernel<<<(NM1 + 255)/256, 256, 0, stream>>>(b1, B1, 360, NM1);
  pad_b_kernel<<<1, 256, 0, stream>>>(b2, B2, 180, NPROJ);
  bias_kernel<<<(6*256*576 + 255)/256, 256, 0, stream>>>(rpi, rpb, BiasT);

  // LN1 -> hA
  ln_kernel<<<ROWS, 64, 0, stream>>>(x, ln1_g, ln1_b, hA);
  // qkv = hA @ Wqkv + b  (bf16)
  gemm_wmma<0><<<dim3(ROWS/128, NQKV/64), 256, 0, stream>>>(
      hA, Wqkv, Bqkv, nullptr, nullptr, QKV, CP, NQKV, 540);
  // attention: one block per (window, head)
  size_t smem = (size_t)16*SLD*4 + 16*32*4 + (size_t)32*KLD*2
              + (size_t)576*32*2 + (size_t)16*SLD*2;
  attn_kernel<<<512*6, 256, smem, stream>>>(QKV, BiasT, Obuf);
  // x2 = o @ proj + b + residual(x)   (f32)
  gemm_wmma<1><<<dim3(ROWS/128, NPROJ/64), 256, 0, stream>>>(
      Obuf, Wproj, Bproj, x, X2, nullptr, CP, NPROJ, C_);
  // LN2 -> H2
  ln_kernel<<<ROWS, 64, 0, stream>>>(X2, ln2_g, ln2_b, H2);
  // m1 = gelu(h2 @ w1 + b1) (bf16)
  gemm_wmma<2><<<dim3(ROWS/128, NM1/64), 256, 0, stream>>>(
      H2, W1, B1, nullptr, nullptr, M1, CP, NM1, 360);
  // out = m1 @ w2 + b2 + x2  (f32)
  gemm_wmma<3><<<dim3(ROWS/128, NPROJ/64), 256, 0, stream>>>(
      M1, W2, B2, X2, out, nullptr, KM2, NPROJ, C_);
}
